// MyGNN2_11355893531404
// MI455X (gfx1250) — matrix operand order repaired and checked by reading, not compiled, not run on hardware
//
#include <hip/hip_runtime.h>

#define WT_STRIDE 132   // bf16 elements; 264B rows -> 8B-aligned b64 LDS loads

typedef __attribute__((ext_vector_type(16))) __bf16 v16bf;
typedef __attribute__((ext_vector_type(2)))  __bf16 v2bf;
typedef __attribute__((ext_vector_type(8)))  float  v8f;

union FragAB { v16bf v; unsigned int u[8]; unsigned short h[16]; };
union BfPair { v2bf v; unsigned int u; unsigned short h[2]; };

__device__ __forceinline__ unsigned int packbf2(float a, float b) {
    BfPair p; p.v[0] = (__bf16)a; p.v[1] = (__bf16)b; return p.u;   // v_cvt_pk_bf16_f32
}
__device__ __forceinline__ unsigned short bf1(float a) {
    union { __bf16 b; unsigned short u; } c; c.b = (__bf16)a; return c.u;
}
__device__ __forceinline__ float lrelu(float v) { return v > 0.f ? v : 0.01f * v; }

// B fragment (32x16 bf16) from LDS: lanes 0-15 K=0..7,16..23, lanes 16-31 K=8..15,24..31.
// bbase is the per-lane base &sWt[ln*WT_STRIDE + hi*8]; (kc,nt) are constants
// after unrolling so offsets fold into the address math.
__device__ __forceinline__ void loadB(FragAB& b, const unsigned short* bbase, int kc, int nt) {
    const unsigned short* bp = bbase + nt * 16 * WT_STRIDE + kc * 32;
    #pragma unroll
    for (int j = 0; j < 4; ++j) {
        uint2 t = *(const uint2*)(bp + (j & 1) * 4 + (j >> 1) * 16);
        b.u[2 * j] = t.x; b.u[2 * j + 1] = t.y;
    }
}

// ---------------------------------------------------------------------------
// GEMM: out[M,128] = act( A[M,128] @ W[128,128] + bias ), bf16 WMMA, f32 acc.
// 256 threads = 8 waves; block covers 128 rows x 128 cols; wave = 16-row slab.
// A fragments preloaded for all K; B fragments double-buffered from LDS so the
// WMMA pipe never waits on dscnt==0.
// ---------------------------------------------------------------------------
template<bool BIAS, bool ACTOUT, bool ACTIN, bool A_BF16, bool OUT_BF16>
__global__ __launch_bounds__(256)
void gemm128(const void* __restrict__ Ain, const float* __restrict__ W,
             const float* __restrict__ bias, void* __restrict__ outp, int M)
{
    __shared__ unsigned short sWt[128 * WT_STRIDE];  // W transposed: Wt[n][k], bf16
    __shared__ float sBias[128];

    const int tid  = threadIdx.x;
    const int lane = tid & 31;
    const int wave = tid >> 5;
    const int hi   = lane >> 4;      // half-wave select
    const int ln   = lane & 15;

    // W (row-major [K=128][N=128]) -> LDS transposed bf16, k-pairs so each pair of
    // HW converts is one v_cvt_pk + one 32-bit LDS store.
    for (int i = tid; i < 64 * 128; i += 256) {
        int n = i & 127, k2 = (i >> 7) << 1;
        BfPair p;
        p.v[0] = (__bf16)W[(size_t)k2 * 128 + n];
        p.v[1] = (__bf16)W[(size_t)(k2 + 1) * 128 + n];
        *(unsigned int*)&sWt[n * WT_STRIDE + k2] = p.u;
    }
    if (BIAS && tid < 128) sBias[tid] = bias[tid];
    __syncthreads();

    const int rowBase = blockIdx.x * 128 + wave * 16;
    int arow = rowBase + ln; if (arow >= M) arow = M - 1;     // clamp; stores guarded
    const float*          arF = (const float*)Ain + (size_t)arow * 128;
    const unsigned short* arB = (const unsigned short*)Ain + (size_t)arow * 128;

    // ---- Preload all 4 A fragments (16x32 bf16 each, layout per ISA 7.12.2) ----
    FragAB a[4];
    #pragma unroll
    for (int kc = 0; kc < 4; ++kc) {
        if (A_BF16) {
            uint4 t0 = *(const uint4*)(arB + kc * 32 + hi * 8);
            uint4 t1 = *(const uint4*)(arB + kc * 32 + hi * 8 + 16);
            a[kc].u[0] = t0.x; a[kc].u[1] = t0.y; a[kc].u[2] = t0.z; a[kc].u[3] = t0.w;
            a[kc].u[4] = t1.x; a[kc].u[5] = t1.y; a[kc].u[6] = t1.z; a[kc].u[7] = t1.w;
        } else {
            #pragma unroll
            for (int p = 0; p < 4; ++p) {
                int k0 = kc * 32 + (p >= 2 ? 16 : 0) + hi * 8 + (p & 1) * 4;
                float4 f = *(const float4*)(arF + k0);
                if (ACTIN) { f.x = lrelu(f.x); f.y = lrelu(f.y); f.z = lrelu(f.z); f.w = lrelu(f.w); }
                a[kc].u[2 * p]     = packbf2(f.x, f.y);
                a[kc].u[2 * p + 1] = packbf2(f.z, f.w);
            }
        }
    }

    v8f acc[8];
    v8f zero = {};
    #pragma unroll
    for (int nt = 0; nt < 8; ++nt) acc[nt] = zero;

    // ---- MAC loop: 32 WMMAs, B double-buffered one iteration ahead ----
    const unsigned short* bbase = &sWt[ln * WT_STRIDE + hi * 8];
    FragAB b[2];
    loadB(b[0], bbase, 0, 0);
    #pragma unroll
    for (int it = 0; it < 32; ++it) {
        const int kc = it >> 3, nt = it & 7;
        if (it + 1 < 32) {
            const int jt = it + 1;
            loadB(b[jt & 1], bbase, jt >> 3, jt & 7);
        }
        acc[nt] = __builtin_amdgcn_wmma_f32_16x16x32_bf16(
            false, a[kc].v, false, b[it & 1].v, (short)0, acc[nt], false, false);
    }

    // ---- Epilogue: C/D layout -> VGPR r holds M = r + hi*8, N = ln per 16-col tile
    float*          outF = (float*)outp;
    unsigned short* outB = (unsigned short*)outp;
    const bool fullBlock = (blockIdx.x * 128 + 128) <= M;   // whole WG in range
    #pragma unroll
    for (int nt = 0; nt < 8; ++nt) {
        int col = nt * 16 + ln;
        float bv = BIAS ? sBias[col] : 0.f;
        if (fullBlock) {
            #pragma unroll
            for (int r = 0; r < 8; ++r) {
                int row = rowBase + r + hi * 8;
                float v = acc[nt][r] + bv;
                if (ACTOUT) v = lrelu(v);
                if (OUT_BF16) outB[(size_t)row * 128 + col] = bf1(v);
                else          outF[(size_t)row * 128 + col] = v;
            }
        } else {
            #pragma unroll
            for (int r = 0; r < 8; ++r) {
                int row = rowBase + r + hi * 8;
                if (row < M) {
                    float v = acc[nt][r] + bv;
                    if (ACTOUT) v = lrelu(v);
                    if (OUT_BF16) outB[(size_t)row * 128 + col] = bf1(v);
                    else          outF[(size_t)row * 128 + col] = v;
                }
            }
        }
    }
}

// ---------------------------------------------------------------------------
// Graph normalization + aggregation kernels
// ---------------------------------------------------------------------------
__global__ void zero_kernel(float* p, int n) {
    int i = blockIdx.x * 256 + threadIdx.x;
    if (i < n) p[i] = 0.f;
}
__global__ void deg_kernel(const int* __restrict__ dst, float* deg, int E) {
    int e = blockIdx.x * 256 + threadIdx.x;
    if (e < E) atomicAdd(&deg[dst[e]], 1.0f);
}
__global__ void dis_kernel(float* d, int n) {
    int i = blockIdx.x * 256 + threadIdx.x;
    if (i < n) d[i] = rsqrtf(d[i] + 1.0f);     // deg of A+I, then ^-1/2
}
// out[i] = h[i] * dis[i]^2 + bias  (self-loop term; fully initializes out)
__global__ void self_init_kernel(const float* __restrict__ h, const float* __restrict__ dis,
                                 const float* __restrict__ bias, float* __restrict__ out, int n) {
    int t = blockIdx.x * 256 + threadIdx.x;
    if (t >= n * 32) return;
    int i = t >> 5, c = (t & 31) << 2;
    float d2 = dis[i] * dis[i];
    float4 hv = *(const float4*)(h + (size_t)i * 128 + c);
    float4 bv = *(const float4*)(bias + c);
    float4 o = { hv.x * d2 + bv.x, hv.y * d2 + bv.y, hv.z * d2 + bv.z, hv.w * d2 + bv.w };
    *(float4*)(out + (size_t)i * 128 + c) = o;
}
// out[dst] += h[src] * dis[src]*dis[dst]   (32 threads / edge, 4 floats each)
__global__ void edge_scatter_kernel(const int* __restrict__ src, const int* __restrict__ dst,
                                    const float* __restrict__ h, const float* __restrict__ dis,
                                    float* out, int E) {
    long long t = (long long)blockIdx.x * 256 + threadIdx.x;
    if (t >= (long long)E * 32) return;
    int e = (int)(t >> 5), c = ((int)t & 31) << 2;
    int s = src[e], d = dst[e];
    float coef = dis[s] * dis[d];
    float4 hv = *(const float4*)(h + (size_t)s * 128 + c);
    float* o = out + (size_t)d * 128 + c;
    atomicAdd(o + 0, hv.x * coef); atomicAdd(o + 1, hv.y * coef);
    atomicAdd(o + 2, hv.z * coef); atomicAdd(o + 3, hv.w * coef);
}
// s[i] = dot( lrelu(h[i,:]), w )   (final 128->1 projection, act fused on input)
__global__ void dot_act_kernel(const float* __restrict__ h, const float* __restrict__ w,
                               float* __restrict__ s, int n) {
    __shared__ float sw[128];
    if (threadIdx.x < 128) sw[threadIdx.x] = w[threadIdx.x];
    __syncthreads();
    int i = blockIdx.x * 256 + threadIdx.x;
    if (i >= n) return;
    const float4* hp = (const float4*)(h + (size_t)i * 128);
    float acc = 0.f;
    #pragma unroll
    for (int j = 0; j < 32; ++j) {
        float4 f = hp[j];
        acc += lrelu(f.x) * sw[4 * j]     + lrelu(f.y) * sw[4 * j + 1]
             + lrelu(f.z) * sw[4 * j + 2] + lrelu(f.w) * sw[4 * j + 3];
    }
    s[i] = acc;
}
__global__ void scalar_init_kernel(const float* __restrict__ s, const float* __restrict__ dis,
                                   const float* __restrict__ bo, float* out, int n) {
    int i = blockIdx.x * 256 + threadIdx.x;
    if (i < n) out[i] = s[i] * dis[i] * dis[i] + bo[0];
}
__global__ void scalar_scatter_kernel(const int* __restrict__ src, const int* __restrict__ dst,
                                      const float* __restrict__ s, const float* __restrict__ dis,
                                      float* out, int E) {
    int e = blockIdx.x * 256 + threadIdx.x;
    if (e < E) atomicAdd(&out[dst[e]], s[src[e]] * dis[src[e]] * dis[dst[e]]);
}

// ---------------------------------------------------------------------------
extern "C" void kernel_launch(void* const* d_in, const int* in_sizes, int n_in,
                              void* d_out, int out_size, void* d_ws, size_t ws_size,
                              hipStream_t stream)
{
    const float* x   = (const float*)d_in[0];
    const int*   ei  = (const int*)d_in[1];
    const float* W1  = (const float*)d_in[2];  const float* b1  = (const float*)d_in[3];
    const float* W2  = (const float*)d_in[4];  const float* b2  = (const float*)d_in[5];
    const float* W3  = (const float*)d_in[6];  const float* b3  = (const float*)d_in[7];
    const float* Wg0 = (const float*)d_in[8];  const float* bg0 = (const float*)d_in[9];
    const float* Wg1 = (const float*)d_in[10]; const float* bg1 = (const float*)d_in[11];
    const float* Wo  = (const float*)d_in[12]; const float* bo  = (const float*)d_in[13];

    const int N = in_sizes[0] / 128;
    const int E = in_sizes[1] / 2;
    const int* src = ei;
    const int* dst = ei + E;

    char*  ws  = (char*)d_ws;
    size_t NBH = (size_t)N * 128 * sizeof(unsigned short);   // bf16 activation buffer
    size_t NBF = (size_t)N * 128 * sizeof(float);            // f32 activation buffer
    void*  bbuf0 = (void*)(ws);                 // bf16
    void*  bbuf1 = (void*)(ws + NBH);           // bf16
    float* fbuf0 = (float*)(ws + 2 * NBH);
    float* fbuf1 = (float*)(ws + 2 * NBH + NBF);
    float* dis   = (float*)(ws + 2 * NBH + 2 * NBF);
    float* svec  = (float*)(ws + 2 * NBH + 2 * NBF + (size_t)N * sizeof(float));
    float* fbuf2 = (float*)(ws);                // aliases bbuf0+bbuf1 (free by then)
    float* outv  = (float*)d_out;

    dim3 blk(256);
    int gN    = (N + 255) / 256;
    int gE    = (E + 255) / 256;
    int gGemm = (N + 127) / 128;
    int gSelf = (N * 32 + 255) / 256;
    int gScat = (int)(((long long)E * 32 + 255) / 256);

    // symmetric-normalization coefficients (shared by all three convs)
    zero_kernel<<<gN, blk, 0, stream>>>(dis, N);
    deg_kernel <<<gE, blk, 0, stream>>>(dst, dis, E);
    dis_kernel <<<gN, blk, 0, stream>>>(dis, N);

    // MLP feature extractor: bias+lrelu fused; intermediates kept in bf16
    gemm128<true,  true,  false, false, true ><<<gGemm, blk, 0, stream>>>(x,     W1, b1, bbuf0, N);
    gemm128<true,  true,  false, true,  true ><<<gGemm, blk, 0, stream>>>(bbuf0, W2, b2, bbuf1, N);
    gemm128<true,  true,  false, true,  true ><<<gGemm, blk, 0, stream>>>(bbuf1, W3, b3, bbuf0, N);

    // GCN conv 0: h2 = h @ Wg0 (bf16 A, f32 out for atomics); lrelu deferred
    gemm128<false, false, false, true,  false><<<gGemm, blk, 0, stream>>>(bbuf0, Wg0, nullptr, fbuf0, N);
    self_init_kernel   <<<gSelf, blk, 0, stream>>>(fbuf0, dis, bg0, fbuf1, N);
    edge_scatter_kernel<<<gScat, blk, 0, stream>>>(src, dst, fbuf0, dis, fbuf1, E);

    // GCN conv 1: lrelu of conv0 output fused into f32 A-load
    gemm128<false, false, true,  false, false><<<gGemm, blk, 0, stream>>>(fbuf1, Wg1, nullptr, fbuf0, N);
    self_init_kernel   <<<gSelf, blk, 0, stream>>>(fbuf0, dis, bg1, fbuf2, N);
    edge_scatter_kernel<<<gScat, blk, 0, stream>>>(src, dst, fbuf0, dis, fbuf2, E);

    // GCN conv 2: 128 -> 1 (lrelu of conv1 output fused into dot)
    dot_act_kernel       <<<gN, blk, 0, stream>>>(fbuf2, Wo, svec, N);
    scalar_init_kernel   <<<gN, blk, 0, stream>>>(svec, dis, bo, outv, N);
    scalar_scatter_kernel<<<gE, blk, 0, stream>>>(src, dst, svec, dis, outv, E);
}
